// ResidualGNNMessagePassing_70128226009226
// MI455X (gfx1250) — matrix operand order repaired and checked by reading, not compiled, not run on hardware
//
#include <hip/hip_runtime.h>

#define HID    128
#define K2     256      // 2*HID
#define TILE_M 16
#define APAD   264      // 256 + 8 bf16 pad -> 528B row stride, 16B aligned, bank-spread

typedef __attribute__((ext_vector_type(8)))  float  v8f;
typedef __attribute__((ext_vector_type(8)))  __bf16 v8bf;
typedef __attribute__((ext_vector_type(16))) __bf16 v16bf;

// ---------------- helpers ----------------

__global__ void zero_out_kernel(float4* __restrict__ out, int n4) {
    int i = blockIdx.x * blockDim.x + threadIdx.x;
    if (i < n4) out[i] = make_float4(0.f, 0.f, 0.f, 0.f);
}

// sws layout (ints): [0..7]=counts  [8..15]=offsets  [16..23]=cursors  [32..32+E)=order
__global__ void init_ws_kernel(int* __restrict__ sws) {
    if (threadIdx.x < 32) sws[threadIdx.x] = 0;
}

__global__ void count_kernel(const int* __restrict__ etype, int* __restrict__ sws, int E) {
    int e = blockIdx.x * blockDim.x + threadIdx.x;
    if (e < E) atomicAdd(&sws[etype[e] - 1], 1);
}

__global__ void scan_kernel(int* __restrict__ sws, int ntypes) {
    if (threadIdx.x == 0) {
        int acc = 0;
        for (int t = 0; t < ntypes; ++t) {
            sws[8 + t]  = acc;   // segment offset
            sws[16 + t] = acc;   // placement cursor
            acc += sws[t];
        }
    }
}

__global__ void place_kernel(const int* __restrict__ etype, int* __restrict__ sws, int E) {
    int e = blockIdx.x * blockDim.x + threadIdx.x;
    if (e < E) {
        int pos = atomicAdd(&sws[16 + (etype[e] - 1)], 1);
        sws[32 + pos] = e;
    }
}

// ---------------- pack W -> bf16 in B-fragment layout ----------------
// dword u = (((t*8 + nb)*8 + ks)*32 + lane)*8 + j
// holds {W[k][n], W[k+1][n]} packed bf16, k = ks*32 + (lane>>4)*16 + 2j,
// n = nb*16 + (lane&15). Each lane's fragment = 32 contiguous bytes.
__global__ void pack_w_kernel(const float* __restrict__ W,
                              unsigned int* __restrict__ Wbf, int nDwords) {
    int u = blockIdx.x * blockDim.x + threadIdx.x;
    if (u >= nDwords) return;
    const int j    = u & 7;
    const int lane = (u >> 3) & 31;
    const int ks   = (u >> 8) & 7;
    const int nb   = (u >> 11) & 7;
    const int t    = u >> 14;
    const int k = ks * 32 + (lane >> 4) * 16 + 2 * j;
    const int n = nb * 16 + (lane & 15);
    const float* Wt = W + (long)t * K2 * HID;
    union { __bf16 b[2]; unsigned int u32; } p;
    p.b[0] = (__bf16)Wt[(k    ) * HID + n];
    p.b[1] = (__bf16)Wt[(k + 1) * HID + n];
    Wbf[u] = p.u32;
}

// ---------------- main WMMA kernel ----------------
// grid = (ceil(E/16), N_TYPES), block = 256 (8 waves)
// block -> 16 edges of one type; wave w -> output columns [16w, 16w+16)

__global__ __launch_bounds__(256) void gnn_wmma_kernel(
    const float*        __restrict__ emb,
    const int*          __restrict__ src,
    const int*          __restrict__ tgt,
    const unsigned int* __restrict__ Wbf,   // packed bf16 fragments
    const int*          __restrict__ sws,
    float*              __restrict__ out)
{
    const int t        = blockIdx.y;
    const int segCount = sws[t];
    const int segStart = sws[8 + t];
    const int tileOff  = blockIdx.x * TILE_M;
    if (tileOff >= segCount) return;                 // block-uniform exit
    const int nval = min(TILE_M, segCount - tileOff);
    const int* order = sws + 32;

    __shared__ __attribute__((aligned(16))) __bf16 sA[TILE_M][APAD];
    __shared__ int s_tgt[TILE_M];

    const int tid = threadIdx.x;
    // -------- stage A tile: 16 edges x 256 cols, f32 -> bf16 --------
    {
        const int r    = tid >> 4;   // edge row 0..15
        const int cg   = tid & 15;   // 16-float chunk 0..15
        const int base = cg * 16;
        if (r < nval) {
            const int e = order[segStart + tileOff + r];
            if (cg == 0) s_tgt[r] = tgt[e];
            const int node   = (cg < 8) ? src[e] : tgt[e];
            const int colOff = (cg & 7) * 16;
            const float4* p  = (const float4*)(emb + (long)node * HID + colOff);
            #pragma unroll
            for (int q = 0; q < 4; ++q) {
                float4 v = p[q];
                sA[r][base + q*4 + 0] = (__bf16)v.x;
                sA[r][base + q*4 + 1] = (__bf16)v.y;
                sA[r][base + q*4 + 2] = (__bf16)v.z;
                sA[r][base + q*4 + 3] = (__bf16)v.w;
            }
        } else {
            if (cg == 0) s_tgt[r] = -1;
            #pragma unroll
            for (int q = 0; q < 16; ++q) sA[r][base + q] = (__bf16)0.0f;
        }
    }
    __syncthreads();

    // -------- per-wave 16x16 output tile, K = 256 in 8 steps of 32 --------
    const int wave = tid >> 5;        // 0..7 -> N slice
    const int lane = tid & 31;
    const int row  = lane & 15;       // A row / B,C column index
    const int hi   = lane >> 4;       // lane half
    const int gcol = wave * 16 + row; // global output column
    // base of this (type, N-block) fragment region, in dwords
    const unsigned int* WbfT = Wbf + ((long)(t * 8 + wave) * 8) * 256 + lane * 8;

    v8f c = {};
    #pragma unroll
    for (int ks = 0; ks < K2 / 32; ++ks) {
        const int kb = ks * 32;
        // A fragment (contiguous K, matches packed B): VGPR0-3 = K kb+hi*16+0..7, VGPR4-7 = K kb+hi*16+8..15
        v8bf alo = *(const v8bf*)&sA[row][kb + hi * 16];
        v8bf ahi = *(const v8bf*)&sA[row][kb + hi * 16 + 8];
        v16bf a;
        #pragma unroll
        for (int i = 0; i < 8; ++i) { a[i] = alo[i]; a[8 + i] = ahi[i]; }

        // B fragment: 32 contiguous bytes per lane, pre-packed bf16 -> two b128 loads
        const v8bf* pb = (const v8bf*)(WbfT + ks * 256);
        v8bf blo = pb[0];
        v8bf bhi = pb[1];
        v16bf b;
        #pragma unroll
        for (int i = 0; i < 8; ++i) { b[i] = blo[i]; b[8 + i] = bhi[i]; }

        c = __builtin_amdgcn_wmma_f32_16x16x32_bf16(
                /*neg_a=*/false, a, /*neg_b=*/false, b,
                /*c_mod=*/(short)0, c, /*reuse_a=*/false, /*reuse_b=*/false);
    }

    // -------- scatter-add: VGPR ri holds row m = ri + hi*8, col gcol --------
    #pragma unroll
    for (int ri = 0; ri < 8; ++ri) {
        const int m    = ri + hi * 8;
        const int node = s_tgt[m];
        if (node >= 0) atomicAdd(&out[(long)node * HID + gcol], c[ri]);
    }
}

// ---------------- launcher ----------------

extern "C" void kernel_launch(void* const* d_in, const int* in_sizes, int n_in,
                              void* d_out, int out_size, void* d_ws, size_t ws_size,
                              hipStream_t stream) {
    const float* emb = (const float*)d_in[0];
    const int*   te  = (const int*)d_in[1];   // [3, E]: etype, src, tgt
    const float* W   = (const float*)d_in[2]; // [T, 256, 128]
    float* out = (float*)d_out;

    const int E       = in_sizes[1] / 3;
    const int n_types = in_sizes[2] / (K2 * HID);
    const int* etype = te;
    const int* src   = te + E;
    const int* tgt   = te + 2 * E;

    // workspace: [0, nWbfDwords) packed bf16 W, then sort structures
    const int nWbfDwords = n_types * K2 * HID / 2;   // 2 bf16 per dword
    unsigned int* Wbf = (unsigned int*)d_ws;
    int* sws = (int*)d_ws + nWbfDwords;

    const int n4 = out_size / 4;   // HID=128 -> divisible
    zero_out_kernel<<<(n4 + 255) / 256, 256, 0, stream>>>((float4*)d_out, n4);
    pack_w_kernel<<<(nWbfDwords + 255) / 256, 256, 0, stream>>>(W, Wbf, nWbfDwords);
    init_ws_kernel<<<1, 32, 0, stream>>>(sws);
    count_kernel<<<(E + 255) / 256, 256, 0, stream>>>(etype, sws, E);
    scan_kernel<<<1, 32, 0, stream>>>(sws, n_types);
    place_kernel<<<(E + 255) / 256, 256, 0, stream>>>(etype, sws, E);

    dim3 grid((E + TILE_M - 1) / TILE_M, n_types);
    gnn_wmma_kernel<<<grid, 256, 0, stream>>>(emb, src, tgt, Wbf, sws, out);
}
